// LocalAffinity_59133109732237
// MI455X (gfx1250) — matrix operand order repaired and checked by reading, not compiled, not run on hardware
//
#include <hip/hip_runtime.h>
#include <stdint.h>

#define H   128
#define W   128
#define HW  (H * W)
#define C   512
#define NC  8
#define PATCH 9
#define PAD 4
#define NOFF 81

typedef __attribute__((ext_vector_type(16))) __bf16        v16bf;
typedef __attribute__((ext_vector_type(8)))  float         v8f;
typedef __attribute__((ext_vector_type(4)))  unsigned int  v4u;
typedef __attribute__((ext_vector_type(8)))  unsigned int  v8u;

union BF16x16 {
    v4u   u[2];
    v16bf v;
};

__device__ __forceinline__ unsigned short f2bf(float f) {
    unsigned int u = __float_as_uint(f);
    u += 0x7FFFu + ((u >> 16) & 1u);   // round-to-nearest-even
    return (unsigned short)(u >> 16);
}

// ---------------------------------------------------------------------------
// Kernel 1a: L2-normalize target features over channels, convert to bf16.
// feat_tar: [HW, C] f32 -> tarN [HW, C] bf16. One wave per pixel.
// ---------------------------------------------------------------------------
__global__ __launch_bounds__(256) void norm_tar_kernel(
    const float* __restrict__ ft, unsigned short* __restrict__ tarN)
{
    const int wave = threadIdx.x >> 5;
    const int lane = threadIdx.x & 31;
    const int p    = blockIdx.x * 8 + wave;

    const float* row = ft + (size_t)p * C;
    float s = 0.f;
    for (int c = lane; c < C; c += 32) {
        float v = row[c];
        s += v * v;
    }
#pragma unroll
    for (int o = 16; o > 0; o >>= 1) s += __shfl_xor(s, o, 32);

    const float inv = 1.f / fmaxf(sqrtf(s), 1e-12f);
    unsigned short* orow = tarN + (size_t)p * C;
    for (int c = lane; c < C; c += 32) orow[c] = f2bf(row[c] * inv);
}

// ---------------------------------------------------------------------------
// Kernel 1b: per-pixel inverse L2 norm of reference features.
// feat_refs: [NC, C, HW] f32. Lanes map to pixels (coalesced over HW).
// ---------------------------------------------------------------------------
__global__ __launch_bounds__(256) void refs_norm_kernel(
    const float* __restrict__ fr, float* __restrict__ invn)
{
    const int b    = blockIdx.y;
    const int px   = blockIdx.x * 32 + (threadIdx.x & 31);
    const int crow = threadIdx.x >> 5;

    const float* base = fr + (size_t)b * C * HW + px;
    float s = 0.f;
    for (int c = crow; c < C; c += 8) {
        float v = base[(size_t)c * HW];
        s += v * v;
    }
    __shared__ float red[8][32];
    red[crow][threadIdx.x & 31] = s;
    __syncthreads();
    if (crow == 0) {
        float t = 0.f;
#pragma unroll
        for (int r = 0; r < 8; ++r) t += red[r][threadIdx.x & 31];
        invn[(size_t)b * HW + px] = 1.f / fmaxf(sqrtf(t), 1e-12f);
    }
}

// ---------------------------------------------------------------------------
// Kernel 1c: transpose refs to pixel-major bf16 with normalization applied.
// [NC, C, HW] f32 -> refsN [NC, HW, C] bf16 via 32x32 LDS tile transpose.
// ---------------------------------------------------------------------------
__global__ __launch_bounds__(256) void refs_tr_kernel(
    const float* __restrict__ fr, const float* __restrict__ invn,
    unsigned short* __restrict__ refsN)
{
    const int b  = blockIdx.z;
    const int c0 = blockIdx.x * 32;
    const int p0 = blockIdx.y * 32;
    const int tx = threadIdx.x;
    const int ty = threadIdx.y;

    __shared__ float tile[32][33];
#pragma unroll
    for (int i = 0; i < 4; ++i) {
        int cr = ty + i * 8;
        tile[cr][tx] = fr[((size_t)b * C + c0 + cr) * HW + p0 + tx];
    }
    __syncthreads();
#pragma unroll
    for (int i = 0; i < 4; ++i) {
        int pr = ty + i * 8;
        int p  = p0 + pr;
        float inv = invn[(size_t)b * HW + p];
        refsN[((size_t)(b * HW + p)) * C + c0 + tx] = f2bf(tile[tx][pr] * inv);
    }
}

// ---------------------------------------------------------------------------
// Kernel 2: local correlation via bf16 WMMA + 81-way softmax.
// Block = 288 threads = 9 waves, one wave per dy in [0,9).
// A[16 x 512] target tile staged to LDS by the Tensor Data Mover (TDM) with
// hardware row padding (1024B data + 16B pad = 1040B rows).
// Per wave: G = A * B over two 16x16 N-tiles (24 neighbor columns), K=512.
// ---------------------------------------------------------------------------
__global__ __launch_bounds__(288) void corr_softmax_kernel(
    const unsigned short* __restrict__ tarN,
    const unsigned short* __restrict__ refsN,
    float* __restrict__ out)
{
    const int x0   = blockIdx.x * 16;
    const int y    = blockIdx.y;
    const int b    = blockIdx.z;
    const int tid  = threadIdx.x;
    const int wave = tid >> 5;            // dy in [0, 9)
    const int lane = tid & 31;

    __shared__ __align__(16) unsigned short As[16][520];  // 1040B rows (16B pad)
    __shared__ float corrT[16][96];                       // 81 used

    // ---- stage A tile: 16 pixels x 512 ch, contiguous 16KB in global ----
#if defined(__gfx1250__)
    if (wave == 0) {
        // Tensor DMA descriptor (D#): 1D tile of 2048 x 8B elements with LDS
        // padding: every 256 DWORDs (1024B) insert 4 DWORDs (16B).
        const unsigned ldsAddr = (unsigned)(size_t)(&As[0][0]);
        const unsigned long long ga =
            (unsigned long long)(size_t)(tarN + (size_t)(y * W + x0) * C);

        v4u g0;
        g0[0] = 1u;                                   // count=1, is_restore=0
        g0[1] = ldsAddr;                              // lds_addr
        g0[2] = (unsigned)ga;                         // global_addr[31:0]
        g0[3] = ((unsigned)(ga >> 32) & 0x01FFFFFFu)  // global_addr[56:32]
              | (2u << 30);                           // type=2 ("image")

        v8u g1;
        g1[0] = (3u << 16)      // data_size = 8B
              | (1u << 20)      // pad_enable
              | (7u << 22)      // pad_interval = 256 DWORDs
              | (3u << 25);     // pad_amount  = 4 DWORDs
        g1[1] = (2048u & 0xFFFFu) << 16;  // tensor_dim0[15:0] @ bits 63:48
        g1[2] = (2048u >> 16)             // tensor_dim0[31:16]
              | (1u << 16);               // tensor_dim1 = 1
        g1[3] = (2048u << 16);            // tile_dim0 = 2048 @ bits 127:112
        g1[4] = 0u;                       // tile_dim1 = tile_dim2 = 0
        g1[5] = 2048u;                    // tensor_dim0_stride
        g1[6] = 0u;
        g1[7] = 0u;

        v4u g2 = { 1u, 1u, 0u, 0u };          // tensor_dim2=1, tensor_dim3=1
        v4u g3 = { 0u, 0x00010000u, 0u, 0u }; // tensor_dim4=1, tile_dim4=0

        asm volatile("tensor_load_to_lds %0, %1, %2, %3"
                     :: "s"(g0), "s"(g1), "s"(g2), "s"(g3)
                     : "memory");
        __builtin_amdgcn_s_wait_tensorcnt(0);
    }
#else
    {
        const v4u* src = (const v4u*)(tarN + (size_t)(y * W + x0) * C);
        for (int t = tid; t < 16 * 64; t += 288) {
            int row = t >> 6;
            int cu  = t & 63;
            *((v4u*)((char*)As + row * 1040) + cu) = src[t];
        }
    }
#endif
    __syncthreads();

    // ---- per-wave GEMM over K ----
    const int  dy       = wave;
    const int  yq       = y + dy - PAD;
    const bool rowValid = (yq >= 0) && (yq < H);
    const int  yqc      = rowValid ? yq : 0;

    const int aRow  = lane & 15;             // A: lane holds row M = lane%16
    const int aKoff = (lane >> 4) << 3;      // A: K sub-chunks {0..7,16..23}/{8..15,24..31}
    const int n     = lane & 15;             // B/D: column within 16
    const int bKoff = (lane >> 4) << 3;      // B: K sub-chunks {0..7,16..23}/{8..15,24..31}

    const int  xA = x0 + n - PAD;            // N-tile 0: j = n
    const int  xB = xA + 16;                 // N-tile 1: j = 16 + n
    const bool v0 = rowValid && (xA >= 0) && (xA < W);
    const bool v1 = rowValid && (xB >= 0) && (xB < W);

    const unsigned short* bbase = refsN + (size_t)(b * HW + yqc * W) * C;

    v8f acc0 = {};
    v8f acc1 = {};
    for (int k0 = 0; k0 < C; k0 += 32) {
        // A fragment from LDS: two 16B chunks at K=k0+aKoff and +16
        BF16x16 a;
        {
            const v4u* ap = (const v4u*)((const char*)As + aRow * 1040 +
                                         (size_t)(k0 + aKoff) * 2);
            a.u[0] = ap[0];
            a.u[1] = ap[2];                   // +32 bytes => +16 elements
        }
        // B fragments from global (channel-contiguous per column), zero OOB
        BF16x16 b0, b1;
        b0.u[0] = 0; b0.u[1] = 0;
        b1.u[0] = 0; b1.u[1] = 0;
        if (v0) {
            const v4u* p = (const v4u*)(bbase + (size_t)xA * C + k0 + bKoff);
            b0.u[0] = p[0]; b0.u[1] = p[2];
            if (k0 + 32 < C)
                __builtin_prefetch(bbase + (size_t)xA * C + k0 + 32 + bKoff, 0, 1);
        }
        if (v1) {
            const v4u* p = (const v4u*)(bbase + (size_t)xB * C + k0 + bKoff);
            b1.u[0] = p[0]; b1.u[1] = p[2];
            if (k0 + 32 < C)
                __builtin_prefetch(bbase + (size_t)xB * C + k0 + 32 + bKoff, 0, 1);
        }

        acc0 = __builtin_amdgcn_wmma_f32_16x16x32_bf16(
            false, a.v, false, b0.v, (short)0, acc0, false, false);
        acc1 = __builtin_amdgcn_wmma_f32_16x16x32_bf16(
            false, a.v, false, b1.v, (short)0, acc1, false, false);
    }

    // ---- scatter Gram entries to corr LDS: (m, j) -> (pixel m, dx = j - m) ----
    const int dmBase = (lane >> 4) << 3;     // D: VGPR r, lanes>=16 => M += 8
#pragma unroll
    for (int r = 0; r < 8; ++r) {
        int m   = r + dmBase;
        int dx0 = n - m;                     // j = n
        if (dx0 >= 0 && dx0 < PATCH) corrT[m][dy * PATCH + dx0] = acc0[r];
        int dx1 = 16 + n - m;                // j = 16 + n
        if (dx1 >= 0 && dx1 < PATCH) corrT[m][dy * PATCH + dx1] = acc1[r];
    }
    __syncthreads();

    // ---- softmax over 81 offsets: 16 lanes per pixel, shfl reductions ----
    if (tid < 256) {
        const int i = tid >> 4;              // pixel in tile
        const int s = tid & 15;              // lane within 16-group
        float mx = -3.402823466e+38f;
        for (int o = s; o < NOFF; o += 16) mx = fmaxf(mx, corrT[i][o]);
#pragma unroll
        for (int d = 8; d > 0; d >>= 1) mx = fmaxf(mx, __shfl_xor(mx, d, 16));
        float sum = 0.f;
        for (int o = s; o < NOFF; o += 16) sum += __expf(corrT[i][o] - mx);
#pragma unroll
        for (int d = 8; d > 0; d >>= 1) sum += __shfl_xor(sum, d, 16);
        const float is = 1.f / sum;
        float* op = out + (size_t)(b * HW + y * W + x0 + i) * NOFF;
        for (int o = s; o < NOFF; o += 16) op[o] = __expf(corrT[i][o] - mx) * is;
    }
}

// ---------------------------------------------------------------------------
// Launch
// ---------------------------------------------------------------------------
extern "C" void kernel_launch(void* const* d_in, const int* in_sizes, int n_in,
                              void* d_out, int out_size, void* d_ws, size_t ws_size,
                              hipStream_t stream)
{
    const float* feat_tar  = (const float*)d_in[0];   // [16384, 512]
    const float* feat_refs = (const float*)d_in[1];   // [8, 512, 16384]
    // d_in[2] (label_refs) only defines shapes; unused.
    float* out = (float*)d_out;                       // [8, 16384, 81]

    // workspace layout
    unsigned short* tarN  = (unsigned short*)d_ws;                    // 16 MB
    unsigned short* refsN = tarN + (size_t)HW * C;                    // 128 MB
    float*          invn  = (float*)(refsN + (size_t)NC * HW * C);    // 512 KB

    norm_tar_kernel<<<HW / 8, 256, 0, stream>>>(feat_tar, tarN);
    refs_norm_kernel<<<dim3(HW / 32, NC), 256, 0, stream>>>(feat_refs, invn);
    refs_tr_kernel<<<dim3(C / 32, HW / 32, NC), dim3(32, 8), 0, stream>>>(
        feat_refs, invn, refsN);
    corr_softmax_kernel<<<dim3(W / 16, H, NC), 288, 0, stream>>>(tarN, refsN, out);
}
